// JuliaHybridNode_35253091566234
// MI455X (gfx1250) — matrix operand order repaired and checked by reading, not compiled, not run on hardware
//
#include <hip/hip_runtime.h>
#include <math.h>

typedef __attribute__((ext_vector_type(16))) _Float16 v16h;
typedef __attribute__((ext_vector_type(8)))  _Float16 v8h;
typedef __attribute__((ext_vector_type(8)))  float    v8f;

#define DIMX   512
#define TM     64
#define BATCH  65536

// f16 weight workspace offsets (in halfs)
#define WS_SW2 0
#define WS_CW2 262144
#define WS_NW1 524288
#define WS_NW2 1048576

// output offsets (floats), in return-tuple order
#define OFF_CYC   33554432   // 65536*512
#define OFF_CONN  67108864   // 2*65536*512
#define OFF_ISIN  100663296  // 3*65536*512
#define OFF_FRAC  100728832  // + 65536
#define OFF_JP    100794368  // + 65536

// ---------------------------------------------------------------------------
// Prep: convert the four large fp32 weight matrices to f16 in workspace.
// 2.6MB of f16 weights -> resident in the 192MB L2 for the whole main kernel.
// ---------------------------------------------------------------------------
__global__ __launch_bounds__(256) void cvt_w(const float* __restrict__ sW2,
                                             const float* __restrict__ cW2,
                                             const float* __restrict__ nW1,
                                             const float* __restrict__ nW2,
                                             _Float16* __restrict__ wh) {
    int i = blockIdx.x * 256 + threadIdx.x;
    if (i < 262144) {
        wh[WS_SW2 + i] = (_Float16)sW2[i];
        wh[WS_CW2 + i] = (_Float16)cW2[i];
        wh[WS_NW2 + i] = (_Float16)nW2[i];
    }
    if (i < 524288) wh[WS_NW1 + i] = (_Float16)nW1[i];
}

// ---------------------------------------------------------------------------
// Four 16x16 f32 output tiles (M = 0..63) of one 16-column stripe:
// each B-tile load (1KB/wave from L2) is reused by 4 WMMAs -> 4x less
// weight traffic than one-row-tile-per-load.
// A layout (16-bit A 16x32): lane = row (lane&15); halfs 0..7 -> K=kb..kb+7,
// halfs 8..15 -> K=kb+16..kb+23, kb = (lane>>4)*8.
// B layout (32x16): lane = col (lane&15); same K order as A.
// ---------------------------------------------------------------------------
__device__ __forceinline__ void wmma_col(const _Float16* __restrict__ act, int lda,
                                         const _Float16* __restrict__ w, int ldw,
                                         int cbase, int K, int lane, v8f acc[4]) {
    const int r0   = lane & 15;
    const int kA   = (lane >> 4) * 8;
    const int ncol = cbase + (lane & 15);
    const int kB   = (lane >> 4) * 16;
    for (int k = 0; k < K; k += 32) {
        const _Float16* pb = w + ncol * ldw + k + kA;
        v8h blo = *(const v8h*)pb;
        v8h bhi = *(const v8h*)(pb + 16);
        v16h b;
#pragma unroll
        for (int i = 0; i < 8; ++i) { b[i] = blo[i]; b[i + 8] = bhi[i]; }
#pragma unroll
        for (int m = 0; m < 4; ++m) {
            const _Float16* pa = act + (m * 16 + r0) * lda + k + kA;
            v8h alo = *(const v8h*)pa;
            v8h ahi = *(const v8h*)(pa + 16);
            v16h a;
#pragma unroll
            for (int i = 0; i < 8; ++i) { a[i] = alo[i]; a[i + 8] = ahi[i]; }
            acc[m] = __builtin_amdgcn_wmma_f32_16x16x32_f16(
                /*neg_a=*/false, a, /*neg_b=*/false, b,
                /*c_mod=*/(short)0, acc[m], /*reuse_a=*/false, /*reuse_b=*/false);
        }
    }
}

// ---------------------------------------------------------------------------
// Fully fused: julia features -> s/c MLPs -> combined -> n MLP -> reductions.
// Block = 256 threads (8 wave32), TM=64 batch rows per block, ~258KB LDS.
// ---------------------------------------------------------------------------
__global__ __launch_bounds__(256) void julia_fused(
    const float* __restrict__ inp, const float* __restrict__ cc,
    const float* __restrict__ sW1, const float* __restrict__ sb1,
    const float* __restrict__ sb2,
    const float* __restrict__ cW1, const float* __restrict__ cb1,
    const float* __restrict__ cb2,
    const float* __restrict__ nb1, const float* __restrict__ nb2,
    const _Float16* __restrict__ wh,
    float* __restrict__ out)
{
    __shared__ float    feats[TM][8];
    __shared__ _Float16 actA[TM][DIMX];      // s_h, later n_h
    __shared__ _Float16 actB[TM][DIMX];      // c_h, later connection(f16)
    __shared__ _Float16 actC[TM][2 * DIMX];  // combined = [stab | cyc] (f16)

    const int tid  = threadIdx.x;
    const int wave = tid >> 5;
    const int lane = tid & 31;
    const int row0 = blockIdx.x * TM;

    // ---- 1) Julia iteration features (only 8 of 512 are nonzero) ----
    if (tid < TM) {
        const int g = row0 + tid;
        const float x = inp[2 * g], y = inp[2 * g + 1];
        const float cr = cc[0], ci = cc[1];
        float zr = x, zi = y;
        int   n = 100;
        bool  esc = false;
#pragma unroll 1
        for (int i = 0; i < 100; ++i) {
            if (!esc) {
                float nr = zr * zr - zi * zi + cr;
                float ni = 2.f * zr * zi + ci;
                zr = nr; zi = ni;
                if (nr * nr + ni * ni > 4.f) { n = i; esc = true; }
            }
        }
        feats[tid][0] = (float)n * 0.01f;
        feats[tid][1] = sqrtf(zr * zr + zi * zi);
        feats[tid][2] = zr;
        feats[tid][3] = zi;
        feats[tid][4] = x;
        feats[tid][5] = y;
        feats[tid][6] = cr;
        feats[tid][7] = ci;
    }
    if (blockIdx.x == 0 && tid == 0) { out[OFF_JP] = cc[0]; out[OFF_JP + 1] = cc[1]; }
    __syncthreads();

    // ---- 2) layer 1 (effective K=8, scalar FMAs) -> f16 s_h / c_h ----
    for (int idx = tid; idx < TM * DIMX; idx += 256) {
        const int r = idx >> 9, o = idx & (DIMX - 1);
        const float* f   = feats[r];
        const float* w1s = sW1 + o * DIMX;
        const float* w1c = cW1 + o * DIMX;
        float aS = sb1[o], aC = cb1[o];
#pragma unroll
        for (int j = 0; j < 8; ++j) { aS += f[j] * w1s[j]; aC += f[j] * w1c[j]; }
        actA[r][o] = (_Float16)tanhf(aS);
        actB[r][o] = (_Float16)tanhf(aC);
    }
    __syncthreads();

    // ---- 3a) stability layer 2 (K=512 WMMA, 4 M-tiles per B load) ----
    for (int ct = wave; ct < DIMX / 16; ct += 8) {
        const int cbase = ct * 16;
        v8f acc[4] = {v8f{}, v8f{}, v8f{}, v8f{}};
        wmma_col(&actA[0][0], DIMX, wh + WS_SW2, DIMX, cbase, DIMX, lane, acc);
        const int coln = cbase + (lane & 15);
        const float bS = sb2[coln];
#pragma unroll
        for (int m = 0; m < 4; ++m)
#pragma unroll
            for (int v = 0; v < 8; ++v) {
                const int rl = m * 16 + v + ((lane >> 4) << 3);
                const float vs = acc[m][v] + bS;
                out[(row0 + rl) * DIMX + coln] = vs;
                actC[rl][coln] = (_Float16)vs;
            }
    }
    // ---- 3b) cycle layer 2 ----
    for (int ct = wave; ct < DIMX / 16; ct += 8) {
        const int cbase = ct * 16;
        v8f acc[4] = {v8f{}, v8f{}, v8f{}, v8f{}};
        wmma_col(&actB[0][0], DIMX, wh + WS_CW2, DIMX, cbase, DIMX, lane, acc);
        const int coln = cbase + (lane & 15);
        const float bC = cb2[coln];
#pragma unroll
        for (int m = 0; m < 4; ++m)
#pragma unroll
            for (int v = 0; v < 8; ++v) {
                const int rl = m * 16 + v + ((lane >> 4) << 3);
                const float vc = acc[m][v] + bC;
                out[OFF_CYC + (row0 + rl) * DIMX + coln] = vc;
                actC[rl][DIMX + coln] = (_Float16)vc;
            }
    }
    __syncthreads();

    // ---- 4) n hidden: combined(K=1024) WMMA + tanh -> actA (reused) ----
    for (int ct = wave; ct < DIMX / 16; ct += 8) {
        const int cbase = ct * 16;
        v8f acc[4] = {v8f{}, v8f{}, v8f{}, v8f{}};
        wmma_col(&actC[0][0], 2 * DIMX, wh + WS_NW1, 2 * DIMX, cbase, 2 * DIMX, lane, acc);
        const int coln = cbase + (lane & 15);
        const float b = nb1[coln];
#pragma unroll
        for (int m = 0; m < 4; ++m)
#pragma unroll
            for (int v = 0; v < 8; ++v) {
                const int rl = m * 16 + v + ((lane >> 4) << 3);
                actA[rl][coln] = (_Float16)tanhf(acc[m][v] + b);
            }
    }
    __syncthreads();

    // ---- 5) connection (K=512 WMMA); f32 out + f16 copy for reduction ----
    for (int ct = wave; ct < DIMX / 16; ct += 8) {
        const int cbase = ct * 16;
        v8f acc[4] = {v8f{}, v8f{}, v8f{}, v8f{}};
        wmma_col(&actA[0][0], DIMX, wh + WS_NW2, DIMX, cbase, DIMX, lane, acc);
        const int coln = cbase + (lane & 15);
        const float b = nb2[coln];
#pragma unroll
        for (int m = 0; m < 4; ++m)
#pragma unroll
            for (int v = 0; v < 8; ++v) {
                const int rl = m * 16 + v + ((lane >> 4) << 3);
                const float val = acc[m][v] + b;
                out[OFF_CONN + (row0 + rl) * DIMX + coln] = val;
                actB[rl][coln] = (_Float16)val;
            }
    }
    __syncthreads();

    // ---- 6) row means -> is_in_set, fractal_dimension ----
    if (tid < TM) {
        float sS = 0.f, sC = 0.f;
        for (int j = 0; j < DIMX; ++j) {
            sS += (float)actC[tid][j];
            sC += (float)actB[tid][j];
        }
        const int g = row0 + tid;
        out[OFF_ISIN + g] = (sS * (1.f / 512.f) > 0.5f) ? 1.f : 0.f;
        const float m = sC * (1.f / 512.f);
        out[OFF_FRAC + g] = 1.f / (1.f + expf(-m));
    }
}

extern "C" void kernel_launch(void* const* d_in, const int* in_sizes, int n_in,
                              void* d_out, int out_size, void* d_ws, size_t ws_size,
                              hipStream_t stream) {
    (void)in_sizes; (void)n_in; (void)out_size; (void)ws_size;
    const float* inp = (const float*)d_in[0];
    const float* c   = (const float*)d_in[1];
    const float* sW1 = (const float*)d_in[2];
    const float* sb1 = (const float*)d_in[3];
    const float* sW2 = (const float*)d_in[4];
    const float* sb2 = (const float*)d_in[5];
    const float* cW1 = (const float*)d_in[6];
    const float* cb1 = (const float*)d_in[7];
    const float* cW2 = (const float*)d_in[8];
    const float* cb2 = (const float*)d_in[9];
    const float* nW1 = (const float*)d_in[10];
    const float* nb1 = (const float*)d_in[11];
    const float* nW2 = (const float*)d_in[12];
    const float* nb2 = (const float*)d_in[13];
    float*    out = (float*)d_out;
    _Float16* wh  = (_Float16*)d_ws;

    cvt_w<<<2048, 256, 0, stream>>>(sW2, cW2, nW1, nW2, wh);
    julia_fused<<<BATCH / TM, 256, 0, stream>>>(inp, c, sW1, sb1, sb2,
                                                cW1, cb1, cb2, nb1, nb2, wh, out);
}
